// BiLevelRoutingAttention_10144712753889
// MI455X (gfx1250) — compile-verified, matrix operand order checked by reading, not run on hardware
//
#include <hip/hip_runtime.h>
#include <hip/hip_bf16.h>

// ---------------------------------------------------------------------------
// MI455X (gfx1250) BiLevelRoutingAttention
//  - bf16 WMMA (v_wmma_f32_16x16x32_bf16) for all GEMMs, fp32 accumulate
//  - async global->LDS staging (ASYNCcnt) for GEMM tiles when available
//  - Tensor Data Mover (tensor_load_to_lds, TENSORcnt) for Q/K/V staging
//  - ~113 GFLOP total, ~0.35 GB traffic; 192 MB L2 holds bf16 working set
// ---------------------------------------------------------------------------

typedef __attribute__((ext_vector_type(16))) __bf16 bf16x16;
typedef __attribute__((ext_vector_type(8)))  float  f32x8;
typedef int v4i __attribute__((vector_size(16)));   // async b128 payload type

#define NWIN   7
#define NW     49
#define TW     64      // tokens per window (8x8)
#define CCH    512
#define BB     16
#define NHEADS 16
#define HD     32
#define HHW    56
#define NEG_INF (-1000000000.0f)
#define SCALE  0.17677669529663687f   // 32^-0.5

#define AS_GLOBAL __attribute__((address_space(1)))
#define AS_LDS    __attribute__((address_space(3)))

#if defined(__has_builtin)
#  if __has_builtin(__builtin_amdgcn_global_load_async_to_lds_b128) && \
      __has_builtin(__builtin_amdgcn_s_wait_asynccnt)
#    define USE_ASYNC_LDS 1
#  endif
#  if __has_builtin(__builtin_amdgcn_tensor_load_to_lds) && \
      __has_builtin(__builtin_amdgcn_s_wait_tensorcnt)
#    define USE_TDM 1
#  endif
#endif

__device__ __forceinline__ f32x8 f8zero() {
    f32x8 z = {0.f, 0.f, 0.f, 0.f, 0.f, 0.f, 0.f, 0.f};
    return z;
}

__device__ __forceinline__ f32x8 wmma_bf16(bf16x16 a, bf16x16 b, f32x8 c) {
    // (neg_a, A, neg_b, B, c_mod, C, reuse_a, reuse_b)
    return __builtin_amdgcn_wmma_f32_16x16x32_bf16(false, a, false, b, (short)0, c, false, false);
}

#ifdef USE_ASYNC_LDS
// per-lane 16B async copy global -> LDS, tracked by ASYNCcnt
__device__ __forceinline__ void async_ld_b128(void* lds, const void* g) {
    __builtin_amdgcn_global_load_async_to_lds_b128(
        (AS_GLOBAL v4i*)g, (AS_LDS v4i*)lds, 0, 0);
}
#endif

#ifdef USE_TDM
typedef __attribute__((ext_vector_type(4))) unsigned int u32x4;
typedef __attribute__((ext_vector_type(8))) int          i32x8;
typedef __attribute__((ext_vector_type(4))) int          i32x4;

// 1-D TDM tile load: nelems bf16 elements, contiguous in global, into LDS.
// D# per CDNA5 ISA ch.8: group0 = {count/type/lds_addr/global_addr},
// group1 = {data_size=2B, tensor_dim0=tile_dim0=nelems, stride0=nelems}.
__device__ __forceinline__ void tdm_load_1d(const void* g, void* lds, unsigned nelems) {
    const unsigned long long ga = (unsigned long long)(size_t)g;
    const unsigned ldso = (unsigned)(size_t)(AS_LDS void*)lds;   // LDS byte offset
    u32x4 g0 = {
        1u,                                                // count=1, user D#
        ldso,                                              // lds_addr
        (unsigned)(ga & 0xFFFFFFFFull),                    // global_addr[31:0]
        (unsigned)((ga >> 32) & 0x1FFFFFFull) | (2u << 30) // global_addr[56:32] | type=2
    };
    i32x8 g1 = {
        (int)0x00010000u,                     // wg_mask=0, data_size=1 (2 bytes)
        (int)((nelems & 0xFFFFu) << 16),      // tensor_dim0[15:0]
        (int)((nelems >> 16) & 0xFFFFu),      // tensor_dim0[31:16], tensor_dim1[15:0]=0
        (int)((nelems & 0xFFFFu) << 16),      // tile_dim0 = nelems
        0,                                    // tile_dim1=0 (unused), tile_dim2=0
        (int)nelems,                          // tensor_dim0_stride[31:0]
        0, 0
    };
    i32x4 gz = {0, 0, 0, 0};
#if __clang_major__ >= 23
    i32x8 gz8 = {0, 0, 0, 0, 0, 0, 0, 0};
    __builtin_amdgcn_tensor_load_to_lds(g0, g1, gz, gz, gz8, 0);
#else
    __builtin_amdgcn_tensor_load_to_lds(g0, g1, gz, gz, 0);
#endif
}
#endif

// CDNA5 16-bit A/B fragment, row-major source [row][k] with leading dim ld.
// lane<16: row=lane,  K = {kh+0..kh+7, kh+16..kh+23} with kh=0
// lane>=16: row=lane-16, kh=8  (ISA 7.12.2 16-bit 16x32 layout)
__device__ __forceinline__ bf16x16 frag_rm(const __bf16* base, int row0, int ld) {
    const int lane = threadIdx.x & 31;
    const __bf16* p = base + (row0 + (lane & 15)) * ld + ((lane & 16) ? 8 : 0);
    bf16x16 f;
    *(uint4*)&f         = *(const uint4*)p;          // elements 0..7
    *(((uint4*)&f) + 1) = *(const uint4*)(p + 16);   // elements 8..15
    return f;
}

// B-operand fragment from [K][N] row-major storage (gathered, for V in P*V)
__device__ __forceinline__ bf16x16 frag_kn(const __bf16* base, int k0, int n0, int ld) {
    const int lane = threadIdx.x & 31;
    const int nn = n0 + (lane & 15);
    const int kh = (lane & 16) ? 8 : 0;
    bf16x16 f;
#pragma unroll
    for (int i = 0; i < 8; ++i) {
        f[i]     = base[(k0 + kh + i) * ld + nn];
        f[i + 8] = base[(k0 + kh + 16 + i) * ld + nn];
    }
    return f;
}

// ---------------------------------------------------------------------------
// K0: x (B,C,H,W) f32 -> xw (B,NW,T,C) bf16
// ---------------------------------------------------------------------------
__global__ __launch_bounds__(256) void make_xw_kernel(const float* __restrict__ x,
                                                      __bf16* __restrict__ xw) {
    const size_t i = (size_t)blockIdx.x * 256 + threadIdx.x;
    if (i >= (size_t)BB * NW * TW * CCH) return;
    const int c = (int)(i & (CCH - 1));
    size_t r = i >> 9;
    const int t = (int)(r & (TW - 1));
    r >>= 6;
    const int w = (int)(r % NW);
    const int b = (int)(r / NW);
    const int hh  = (w / NWIN) * 8 + (t >> 3);
    const int ww2 = (w % NWIN) * 8 + (t & 7);
    xw[i] = (__bf16)x[(((size_t)b * CCH + c) * HHW + hh) * HHW + ww2];
}

// ---------------------------------------------------------------------------
// K1: weights f32 -> bf16
// ---------------------------------------------------------------------------
__global__ __launch_bounds__(256) void cvt_w_kernel(const float* __restrict__ qkvw,
                                                    const float* __restrict__ projw,
                                                    __bf16* __restrict__ qkvw_b,
                                                    __bf16* __restrict__ projw_b) {
    const int i = blockIdx.x * 256 + threadIdx.x;
    if (i < 3 * CCH * CCH) qkvw_b[i] = (__bf16)qkvw[i];
    if (i < CCH * CCH)     projw_b[i] = (__bf16)projw[i];
}

// ---------------------------------------------------------------------------
// K2: per-window mean pool: xpool (B,NW,C) f32
// ---------------------------------------------------------------------------
__global__ __launch_bounds__(256) void pool_kernel(const float* __restrict__ x,
                                                   float* __restrict__ xpool) {
    const int bw = blockIdx.x;
    const int b = bw / NW, w = bw % NW;
    const int wr = (w / NWIN) * 8, wc = (w % NWIN) * 8;
    for (int c = threadIdx.x; c < CCH; c += 256) {
        const float* xc = x + ((size_t)b * CCH + c) * (HHW * HHW);
        float s = 0.f;
#pragma unroll
        for (int i = 0; i < 8; ++i)
#pragma unroll
            for (int j = 0; j < 8; ++j)
                s += xc[(wr + i) * HHW + wc + j];
        xpool[((size_t)b * NW + w) * CCH + c] = s * (1.0f / 64.0f);
    }
}

// ---------------------------------------------------------------------------
// K3: router logits + top-4 -> mask (B,NW,NW) f32 {0,1}
//     (top-k of softmax == top-k of logits; softmax skipped)
// ---------------------------------------------------------------------------
__global__ __launch_bounds__(64) void router_kernel(const float* __restrict__ xpool,
                                                    const float* __restrict__ rw,
                                                    const float* __restrict__ rb,
                                                    float* __restrict__ mask) {
    __shared__ float sS[NW];
    __shared__ float sM[NW];
    const int bw = blockIdx.x;
    const int b = bw / NW, w = bw % NW;
    const int k = threadIdx.x;
    if (k < NW) {
        const float* xp = xpool + ((size_t)b * NW + w) * CCH;
        const float* wk = rw + (size_t)k * CCH;
        float s = rb[k];
        for (int c = 0; c < CCH; ++c) s += xp[c] * wk[c];
        sS[k] = s;
        sM[k] = 0.f;
    }
    __syncthreads();
    if (k == 0) {
        for (int it = 0; it < 4; ++it) {
            int best = 0;
            float bv = -3.4e38f;
            for (int j = 0; j < NW; ++j)
                if (sM[j] == 0.f && sS[j] > bv) { bv = sS[j]; best = j; }
            sM[best] = 1.f;
        }
    }
    __syncthreads();
    if (k < NW) mask[((size_t)b * NW + w) * NW + k] = sM[k];
}

// ---------------------------------------------------------------------------
// K4: QKV GEMM  (50176 x 512) x (512 x 1536), bf16 WMMA, fp32 acc.
//     grid.x = window (M tile = 64 tokens), grid.y = 12 (N tile = 128)
//     scatter into Q/K/V layout [B][H][T][NW][HD] (attention-friendly)
// ---------------------------------------------------------------------------
__global__ __launch_bounds__(256) void qkv_gemm_kernel(const __bf16* __restrict__ xw,
                                                       const __bf16* __restrict__ wq,
                                                       __bf16* __restrict__ qb,
                                                       __bf16* __restrict__ kb,
                                                       __bf16* __restrict__ vb) {
    __shared__ __align__(16) __bf16 As[64][32];
    __shared__ __align__(16) __bf16 Bs[128][32];
    const int blkM = blockIdx.x;
    const int n0   = blockIdx.y * 128;
    const int tid  = threadIdx.x;
    const int lane = tid & 31;
    const int wv   = tid >> 5;     // 8 waves
    const int msub = wv & 3;       // 4 x 16-row subtiles
    const int nsub = wv >> 2;      // 2 x 64-col subtiles

    const __bf16* Arow = xw + (size_t)blkM * TW * CCH;
    f32x8 acc[4];
#pragma unroll
    for (int j = 0; j < 4; ++j) acc[j] = f8zero();

    for (int kk = 0; kk < CCH; kk += 32) {
        {
            const int row = tid >> 2, cx = (tid & 3) * 8;
#ifdef USE_ASYNC_LDS
            async_ld_b128(&As[row][cx], Arow + row * CCH + kk + cx);
#pragma unroll
            for (int u = 0; u < 2; ++u) {
                const int ch = tid + u * 256;
                const int rB = ch >> 2, cB = (ch & 3) * 8;
                async_ld_b128(&Bs[rB][cB], wq + (size_t)(n0 + rB) * CCH + kk + cB);
            }
#else
            *(uint4*)&As[row][cx] = *(const uint4*)(Arow + row * CCH + kk + cx);
#pragma unroll
            for (int u = 0; u < 2; ++u) {
                const int ch = tid + u * 256;
                const int rB = ch >> 2, cB = (ch & 3) * 8;
                *(uint4*)&Bs[rB][cB] =
                    *(const uint4*)(wq + (size_t)(n0 + rB) * CCH + kk + cB);
            }
#endif
            if (kk + 32 < CCH)
                __builtin_prefetch(Arow + row * CCH + kk + 32 + cx, 0, 1);
        }
#ifdef USE_ASYNC_LDS
        __builtin_amdgcn_s_wait_asynccnt(0);
#endif
        __syncthreads();
        bf16x16 a = frag_rm(&As[0][0], msub * 16, 32);
#pragma unroll
        for (int j = 0; j < 4; ++j) {
            bf16x16 b = frag_rm(&Bs[0][0], nsub * 64 + j * 16, 32);
            acc[j] = wmma_bf16(a, b, acc[j]);
        }
        __syncthreads();
    }

    const int b   = blkM / NW;
    const int w   = blkM % NW;
    const int col = lane & 15;
    const int rb8 = (lane & 16) ? 8 : 0;
#pragma unroll
    for (int j = 0; j < 4; ++j) {
        const int ng    = n0 + nsub * 64 + j * 16 + col;
        const int three = ng >> 9;
        const int head  = (ng >> 5) & 15;
        const int dd    = ng & 31;
        __bf16* dst = (three == 0) ? qb : (three == 1) ? kb : vb;
#pragma unroll
        for (int r = 0; r < 8; ++r) {
            const int t = msub * 16 + rb8 + r;
            dst[((((size_t)b * NHEADS + head) * TW + t) * NW + w) * HD + dd] =
                (__bf16)acc[j][r];
        }
    }
}

// ---------------------------------------------------------------------------
// K5: attention. One block per (b,h,t); 128 threads (4 waves).
//     Q/K/V staged via TDM (tensor_load_to_lds) when available.
//     S = Q*K^T (pad 49->64), softmax with RPB + routing mask, O = P*V.
// ---------------------------------------------------------------------------
__global__ __launch_bounds__(128) void attn_kernel(const __bf16* __restrict__ qb,
                                                   const __bf16* __restrict__ kb,
                                                   const __bf16* __restrict__ vb,
                                                   const float* __restrict__ mask,
                                                   const float* __restrict__ rpb,
                                                   __bf16* __restrict__ ob) {
    __shared__ __align__(16) __bf16 Qs[64][32];
    __shared__ __align__(16) __bf16 Ks[64][32];
    __shared__ __align__(16) __bf16 Vs[64][32];
    __shared__ __align__(16) float  Ss[64][68];
    __shared__ __align__(16) __bf16 Ps[64][64];

    const int t = blockIdx.x, h = blockIdx.y, b = blockIdx.z;
    const int tid = threadIdx.x;
    const int lane = tid & 31, wv = tid >> 5;

    const size_t base = (((size_t)b * NHEADS + h) * TW + t) * (NW * HD);
    const uint4 z4 = {0u, 0u, 0u, 0u};

#ifdef USE_TDM
    // zero-fill pad rows 49..63 of Q/K/V: 3 mats x 15 rows x 4 chunks = 180
    for (int i = tid; i < 180; i += 128) {
        const int mat = i / 60, ch = i % 60;
        const int row = NW + (ch >> 2), cx = (ch & 3) * 8;
        __bf16* dst = (mat == 0) ? &Qs[0][0] : (mat == 1) ? &Ks[0][0] : &Vs[0][0];
        *(uint4*)(dst + row * 32 + cx) = z4;
    }
    // one TDM descriptor per matrix, issued by distinct waves (uniform branch)
    if (wv == 0)      tdm_load_1d(qb + base, &Qs[0][0], NW * HD);
    else if (wv == 1) tdm_load_1d(kb + base, &Ks[0][0], NW * HD);
    else if (wv == 2) tdm_load_1d(vb + base, &Vs[0][0], NW * HD);
    __builtin_amdgcn_s_wait_tensorcnt(0);
    __syncthreads();
#else
    // stage Q/K/V (49x32 each, zero-pad rows to 64)
    for (int i = tid; i < 768; i += 128) {
        const int mat = i >> 8;
        const int ch  = i & 255;
        const int row = ch >> 2, cx = (ch & 3) * 8;
        const __bf16* src = (mat == 0) ? qb : (mat == 1) ? kb : vb;
        uint4 val = z4;
        if (row < NW) val = *(const uint4*)(src + base + (size_t)ch * 8);
        __bf16* dst = (mat == 0) ? &Qs[0][0] : (mat == 1) ? &Ks[0][0] : &Vs[0][0];
        *(uint4*)(dst + row * 32 + cx) = val;
    }
    __syncthreads();
#endif

    // S = Q * K^T : wave wv owns M-tile wv; K^T fragment == Ks rows as B^T
    {
        bf16x16 aq = frag_rm(&Qs[0][0], wv * 16, 32);
        const int col = lane & 15, rb8 = (lane & 16) ? 8 : 0;
#pragma unroll
        for (int nt = 0; nt < 4; ++nt) {
            bf16x16 bk = frag_rm(&Ks[0][0], nt * 16, 32);
            f32x8 s = f8zero();
            s = wmma_bf16(aq, bk, s);
#pragma unroll
            for (int r = 0; r < 8; ++r)
                Ss[wv * 16 + rb8 + r][nt * 16 + col] = s[r];
        }
    }
    __syncthreads();

    // softmax over k-windows with relative position bias + routing mask
    if (tid < 64) {
        const int m  = tid;
        const int mc = (m < NW) ? m : NW - 1;     // padded rows: harmless dummies
        const float* mrow = mask + ((size_t)b * NW + mc) * NW;
        const int mi = mc / NWIN, mj = mc % NWIN;
        float mx = -3.0e38f;
        for (int k = 0; k < NW; ++k) {
            const int relv = (mi - k / NWIN + 6) * 13 + (mj - k % NWIN + 6);
            const float val = (mrow[k] > 0.5f)
                            ? (Ss[m][k] * SCALE + rpb[relv * NHEADS + h])
                            : NEG_INF;
            Ss[m][k] = val;
            mx = fmaxf(mx, val);
        }
        float sum = 0.f;
        for (int k = 0; k < NW; ++k) {
            const float e = __expf(Ss[m][k] - mx);
            sum += e;
            Ss[m][k] = e;
        }
        const float inv = 1.0f / sum;
        for (int k = 0; k < 64; ++k)
            Ps[m][k] = (k < NW) ? (__bf16)(Ss[m][k] * inv) : (__bf16)0.0f;
    }
    __syncthreads();

    // O = P * V : M=64 (4 waves x 16), N=32 (2 tiles), K=64 (2 steps)
    f32x8 o0 = f8zero(), o1 = f8zero();
#pragma unroll
    for (int ks = 0; ks < 64; ks += 32) {
        bf16x16 ap  = frag_rm(&Ps[0][0] + ks, wv * 16, 64);
        bf16x16 bv0 = frag_kn(&Vs[0][0], ks, 0, 32);
        bf16x16 bv1 = frag_kn(&Vs[0][0], ks, 16, 32);
        o0 = wmma_bf16(ap, bv0, o0);
        o1 = wmma_bf16(ap, bv1, o1);
    }
    {
        const int col = lane & 15, rb8 = (lane & 16) ? 8 : 0;
#pragma unroll
        for (int r = 0; r < 8; ++r) {
            const int m = wv * 16 + rb8 + r;
            if (m < NW) {
                const size_t di = (((size_t)b * NW + m) * TW + t) * CCH + h * HD;
                ob[di + col]      = (__bf16)o0[r];
                ob[di + 16 + col] = (__bf16)o1[r];
            }
        }
    }
}

// ---------------------------------------------------------------------------
// K6: output projection GEMM (50176x512)x(512x512) + bias -> d_out (B,C,H,W) f32
// ---------------------------------------------------------------------------
__global__ __launch_bounds__(256) void proj_gemm_kernel(const __bf16* __restrict__ ob,
                                                        const __bf16* __restrict__ wp,
                                                        const float* __restrict__ pb,
                                                        float* __restrict__ out) {
    __shared__ __align__(16) __bf16 As[64][32];
    __shared__ __align__(16) __bf16 Bs[128][32];
    const int blkM = blockIdx.x;
    const int n0   = blockIdx.y * 128;
    const int tid  = threadIdx.x;
    const int lane = tid & 31;
    const int wv   = tid >> 5;
    const int msub = wv & 3;
    const int nsub = wv >> 2;

    const __bf16* Arow = ob + (size_t)blkM * TW * CCH;
    f32x8 acc[4];
#pragma unroll
    for (int j = 0; j < 4; ++j) acc[j] = f8zero();

    for (int kk = 0; kk < CCH; kk += 32) {
        {
            const int row = tid >> 2, cx = (tid & 3) * 8;
#ifdef USE_ASYNC_LDS
            async_ld_b128(&As[row][cx], Arow + row * CCH + kk + cx);
#pragma unroll
            for (int u = 0; u < 2; ++u) {
                const int ch = tid + u * 256;
                const int rB = ch >> 2, cB = (ch & 3) * 8;
                async_ld_b128(&Bs[rB][cB], wp + (size_t)(n0 + rB) * CCH + kk + cB);
            }
#else
            *(uint4*)&As[row][cx] = *(const uint4*)(Arow + row * CCH + kk + cx);
#pragma unroll
            for (int u = 0; u < 2; ++u) {
                const int ch = tid + u * 256;
                const int rB = ch >> 2, cB = (ch & 3) * 8;
                *(uint4*)&Bs[rB][cB] =
                    *(const uint4*)(wp + (size_t)(n0 + rB) * CCH + kk + cB);
            }
#endif
            if (kk + 32 < CCH)
                __builtin_prefetch(Arow + row * CCH + kk + 32 + cx, 0, 1);
        }
#ifdef USE_ASYNC_LDS
        __builtin_amdgcn_s_wait_asynccnt(0);
#endif
        __syncthreads();
        bf16x16 a = frag_rm(&As[0][0], msub * 16, 32);
#pragma unroll
        for (int j = 0; j < 4; ++j) {
            bf16x16 b = frag_rm(&Bs[0][0], nsub * 64 + j * 16, 32);
            acc[j] = wmma_bf16(a, b, acc[j]);
        }
        __syncthreads();
    }

    const int b  = blkM / NW;
    const int w  = blkM % NW;
    const int wr = (w / NWIN) * 8, wc = (w % NWIN) * 8;
    const int col = lane & 15;
    const int rb8 = (lane & 16) ? 8 : 0;
#pragma unroll
    for (int j = 0; j < 4; ++j) {
        const int d = n0 + nsub * 64 + j * 16 + col;
        const float bias = pb[d];
#pragma unroll
        for (int r = 0; r < 8; ++r) {
            const int tt  = msub * 16 + rb8 + r;
            const int hh  = wr + (tt >> 3);
            const int ww2 = wc + (tt & 7);
            out[(((size_t)b * CCH + d) * HHW + hh) * HHW + ww2] = acc[j][r] + bias;
        }
    }
}

// ---------------------------------------------------------------------------
extern "C" void kernel_launch(void* const* d_in, const int* in_sizes, int n_in,
                              void* d_out, int out_size, void* d_ws, size_t ws_size,
                              hipStream_t stream) {
    const float* x    = (const float*)d_in[0];
    const float* rw   = (const float*)d_in[1];
    const float* rb   = (const float*)d_in[2];
    const float* qkvw = (const float*)d_in[3];
    const float* pw   = (const float*)d_in[4];
    const float* pb   = (const float*)d_in[5];
    const float* rpb  = (const float*)d_in[6];
    float* out = (float*)d_out;

    char* ws = (char*)d_ws;
    size_t off = 0;
    auto alloc = [&](size_t bytes) -> char* {
        char* p = ws + off;
        off = (off + bytes + 255) & ~(size_t)255;
        return p;
    };

    const size_t TOKELEMS = (size_t)BB * NW * TW * CCH;          // 25,690,112
    __bf16* xw      = (__bf16*)alloc(TOKELEMS * 2);              // later reused as O
    __bf16* qkvw_b  = (__bf16*)alloc((size_t)3 * CCH * CCH * 2);
    __bf16* projw_b = (__bf16*)alloc((size_t)CCH * CCH * 2);
    __bf16* qB      = (__bf16*)alloc(TOKELEMS * 2);
    __bf16* kB      = (__bf16*)alloc(TOKELEMS * 2);
    __bf16* vB      = (__bf16*)alloc(TOKELEMS * 2);
    float*  xpool   = (float*)alloc((size_t)BB * NW * CCH * 4);
    float*  maskb   = (float*)alloc((size_t)BB * NW * NW * 4);

    // 1) layout transforms / precision conversion
    make_xw_kernel<<<(unsigned)(TOKELEMS / 256), 256, 0, stream>>>(x, xw);
    cvt_w_kernel<<<(3 * CCH * CCH + 255) / 256, 256, 0, stream>>>(qkvw, pw, qkvw_b, projw_b);

    // 2) router: pool -> logits -> top-4 mask
    pool_kernel<<<BB * NW, 256, 0, stream>>>(x, xpool);
    router_kernel<<<BB * NW, 64, 0, stream>>>(xpool, rw, rb, maskb);

    // 3) QKV projection (bf16 WMMA, async LDS staging)
    qkv_gemm_kernel<<<dim3(BB * NW, 12), 256, 0, stream>>>(xw, qkvw_b, qB, kB, vB);

    // 4) cross-window attention per (b,h,t); writes O into xw buffer (now dead)
    attn_kernel<<<dim3(TW, NHEADS, BB), 128, 0, stream>>>(qB, kB, vB, maskb, rpb, xw);

    // 5) output projection + bias -> (B,C,H,W) fp32
    proj_gemm_kernel<<<dim3(BB * NW, 4), 256, 0, stream>>>(xw, projw_b, pb, out);
}
